// FlaxLLaMAAttention_5102421148364
// MI455X (gfx1250) — matrix operand order repaired and checked by reading, not compiled, not run on hardware
//
#include <hip/hip_runtime.h>
#include <hip/hip_bf16.h>

// ---------------------------------------------------------------------------
// FlaxLLaMAAttention forward for MI455X (gfx1250, wave32, WMMA).
// B=1, S=2048, HID=2048, H=16, KV=4, D=128, RoPE theta=10000, causal GQA.
// fp32 -> f16 once; all GEMMs via v_wmma_f32_16x16x32_f16 (f32 accumulate);
// flash-attention tiles in LDS/VGPRs; fp32 softmax with wave32 shuffles.
// CDNA5 data movement: global_load_async_to_lds_b128 (ASYNCcnt) streams the
// GEMM A-tiles and attention K-tiles straight into LDS, double-buffered so
// WMMA on tile k overlaps the copy of tile k+1.
// ---------------------------------------------------------------------------

typedef _Float16 v8h  __attribute__((ext_vector_type(8)));
typedef _Float16 v16h __attribute__((ext_vector_type(16)));
typedef float    v8f  __attribute__((ext_vector_type(8)));

#define S_LEN 2048
#define HID   2048
#define NH    16
#define NKV   4
#define DH    128

static __device__ inline v8f wmma_f16(v16h a, v16h b, v8f c) {
  return __builtin_amdgcn_wmma_f32_16x16x32_f16(
      /*neg_a=*/false, a, /*neg_b=*/false, b,
      /*c_mod=*/(short)0, c, /*reuse_a=*/false, /*reuse_b=*/false);
}

// CDNA5 async global->LDS copy (16B per lane), tracked by ASYNCcnt.
// VDST carries the LDS byte address (low 32 bits of the generic pointer).
static __device__ inline void async_copy_b128(void* lds_dst, const void* gsrc) {
  unsigned int lds_off = (unsigned int)(unsigned long long)lds_dst;
  asm volatile("global_load_async_to_lds_b128 %0, %1, off"
               :
               : "v"(lds_off), "v"(gsrc)
               : "memory");
}
static __device__ inline void wait_async0() {
  asm volatile("s_wait_asynccnt 0x0" ::: "memory");
}

// Build a 16x32 f16 A-fragment from a row-major row pointer.
// Lanes 0-15 (M=lane): halves 0..15 = K koff+0..15;
// lanes 16-31 use koff += 16 (folded into koff by caller).
static __device__ inline v16h frag_a_row(const _Float16* row, int koff) {
  v8h lo = *(const v8h*)(row + koff);
  v8h hi = *(const v8h*)(row + koff + 8);
  v16h r;
#pragma unroll
  for (int i = 0; i < 8; ++i) { r[i] = lo[i]; r[8 + i] = hi[i]; }
  return r;
}

// ---------------------------------------------------------------------------
// fp32 -> f16 convert
// ---------------------------------------------------------------------------
__global__ void cvt_f32_to_f16(const float* __restrict__ in,
                               _Float16* __restrict__ out, int n) {
  int i = blockIdx.x * 256 + threadIdx.x;
  if (i < n) out[i] = (_Float16)in[i];
}

// ---------------------------------------------------------------------------
// RoPE on fp32 GEMM output, emit f16.  x: [S, nh, 128]
// ---------------------------------------------------------------------------
__global__ void rope_cvt(const float* __restrict__ x,
                         const int* __restrict__ pos_ids,
                         _Float16* __restrict__ out, int nh, int npairs) {
  int i = blockIdx.x * 256 + threadIdx.x;
  if (i >= npairs) return;
  int j    = i & 63;           // pair index within head dim (DH/2 = 64)
  int rest = i >> 6;           // s * nh + head
  int s    = rest / nh;
  size_t base = (size_t)rest * DH + 2 * j;
  float x0 = x[base], x1 = x[base + 1];
  float pos  = (float)pos_ids[s];
  float freq = __powf(10000.0f, -(float)(2 * j) / (float)DH);
  float ang  = pos * freq;
  float c = __cosf(ang), sn = __sinf(ang);
  out[base]     = (_Float16)(x0 * c - x1 * sn);
  out[base + 1] = (_Float16)(x0 * sn + x1 * c);
}

// ---------------------------------------------------------------------------
// Generic f16 GEMM: C[M,N] = A[M,K] * B[K,N], row-major, f32 accumulate.
// 128 threads = 4 waves; 128x128 C tile; each wave 64x64 (4x4 WMMA tiles).
// Double-buffered LDS: A streamed via async global->LDS; B software-pipelined
// through VGPRs (needs transpose on store).  M,N mult of 128; K mult of 32.
// ---------------------------------------------------------------------------
__global__ __launch_bounds__(128) void gemm_f16(
    const _Float16* __restrict__ A, const _Float16* __restrict__ B,
    float* __restrict__ Cf, _Float16* __restrict__ Ch, int M, int N, int K) {
  __shared__ alignas(32) _Float16 lA[2][128][32];   // [buf][m][k]
  __shared__ alignas(32) _Float16 lBt[2][128][32];  // [buf][n][k]

  int t = threadIdx.x;
  int lane = t & 31, wave = t >> 5;
  int lm = lane & 15, lh = lane >> 4;
  int m0b = blockIdx.y * 128, n0b = blockIdx.x * 128;
  int waveM = (wave >> 1) * 64, waveN = (wave & 1) * 64;

  v8f z = {0.f, 0.f, 0.f, 0.f, 0.f, 0.f, 0.f, 0.f};
  v8f acc[4][4];
#pragma unroll
  for (int i = 0; i < 4; ++i)
#pragma unroll
    for (int j = 0; j < 4; ++j) acc[i][j] = z;

  // Per-thread tile-chunk coordinates (4 x 16B chunks each for A and B)
  v8h breg[4];

  // --- prologue: issue async A copy for k0=0, load B chunks for k0=0 ---
#pragma unroll
  for (int c4 = 0; c4 < 4; ++c4) {
    int c = t + c4 * 128;
    int r = c >> 2, col = (c & 3) * 8;  // A: 128 rows x 32 halves
    async_copy_b128(&lA[0][r][col], &A[(size_t)(m0b + r) * K + col]);
  }
#pragma unroll
  for (int c4 = 0; c4 < 4; ++c4) {
    int c = t + c4 * 128;
    int r = c >> 4, col = (c & 15) * 8;  // B: 32 rows x 128 halves
    breg[c4] = *(const v8h*)&B[(size_t)r * N + n0b + col];
  }

  const int nk = K >> 5;
  for (int ki = 0; ki < nk; ++ki) {
    int buf = ki & 1;
    // Commit the prefetched B chunks (transposed) into this iteration's buffer
#pragma unroll
    for (int c4 = 0; c4 < 4; ++c4) {
      int c = t + c4 * 128;
      int r = c >> 4, col = (c & 15) * 8;
#pragma unroll
      for (int e = 0; e < 8; ++e) lBt[buf][col + e][r] = breg[c4][e];
    }
    wait_async0();     // A tile for this iteration has landed in LDS
    __syncthreads();   // all waves' A-copies and B-stores visible

    if (ki + 1 < nk) {
      int k0n = (ki + 1) << 5;
      // Stream next A tile into the other buffer (overlaps WMMA below)
#pragma unroll
      for (int c4 = 0; c4 < 4; ++c4) {
        int c = t + c4 * 128;
        int r = c >> 2, col = (c & 3) * 8;
        async_copy_b128(&lA[buf ^ 1][r][col],
                        &A[(size_t)(m0b + r) * K + k0n + col]);
      }
      // Prefetch next B tile into registers (overlaps WMMA below)
#pragma unroll
      for (int c4 = 0; c4 < 4; ++c4) {
        int c = t + c4 * 128;
        int r = c >> 4, col = (c & 15) * 8;
        breg[c4] = *(const v8h*)&B[(size_t)(k0n + r) * N + n0b + col];
      }
    }

    // ---- compute: 4x4 WMMA tiles over this 32-wide K slice ----
    v16h af[4];
    int koffA = lh * 16;
#pragma unroll
    for (int i = 0; i < 4; ++i)
      af[i] = frag_a_row(&lA[buf][waveM + i * 16 + lm][0], koffA);
    int kstartB = lh * 16;  // B frag: lanes 0-15 K=0..15, lanes 16-31 K=16..31
#pragma unroll
    for (int j = 0; j < 4; ++j) {
      v16h bf = *(const v16h*)&lBt[buf][waveN + j * 16 + lm][kstartB];
#pragma unroll
      for (int i = 0; i < 4; ++i) acc[i][j] = wmma_f16(af[i], bf, acc[i][j]);
    }
    __syncthreads();  // done reading buf before it is overwritten again
  }

  // Store: C layout — lane lm = column, lh*8+v = row within each 16x16 tile
  int rbase = m0b + waveM + lh * 8;
  int cbase = n0b + waveN + lm;
#pragma unroll
  for (int i = 0; i < 4; ++i)
#pragma unroll
    for (int j = 0; j < 4; ++j)
#pragma unroll
      for (int v = 0; v < 8; ++v) {
        size_t idx = (size_t)(rbase + i * 16 + v) * N + cbase + j * 16;
        if (Cf) Cf[idx] = acc[i][j][v];
        else    Ch[idx] = (_Float16)acc[i][j][v];
      }
}

// ---------------------------------------------------------------------------
// Flash attention, one (head, 64-row q tile) per block.  128 threads = 4
// waves; wave w owns q rows [qb*64 + w*16, +16).  KV tiles of 64 iterated
// causally.  GQA: kv head = h >> 2.  K tile streamed async global->LDS.
// ---------------------------------------------------------------------------
__global__ __launch_bounds__(128) void attn_flash(
    const _Float16* __restrict__ Qh, const _Float16* __restrict__ Kh,
    const _Float16* __restrict__ Vh, const int* __restrict__ amask,
    _Float16* __restrict__ Oh) {
  __shared__ alignas(32) _Float16 lK[64][128];    // [kv][d]  == B[n][k] for QK^T
  __shared__ alignas(32) _Float16 lVt[128][64];   // [d][kv]  == B[n][k] for P@V
  __shared__ alignas(32) _Float16 lP[4][16][64];  // per-wave P staging

  int t = threadIdx.x, lane = t & 31, wave = t >> 5;
  int lm = lane & 15, lh = lane >> 4;
  int qb = blockIdx.x, h = blockIdx.y, kvh = h >> 2;
  int qrow0 = qb * 64 + wave * 16;

  // Q fragments for this wave's 16 rows (resident for whole kernel)
  v16h qf[4];
  {
    const _Float16* qrow = Qh + (size_t)(qrow0 + lm) * (NH * DH) + h * DH;
#pragma unroll
    for (int kc = 0; kc < 4; ++kc) qf[kc] = frag_a_row(qrow, kc * 32 + lh * 16);
  }

  v8f z = {0.f, 0.f, 0.f, 0.f, 0.f, 0.f, 0.f, 0.f};
  v8f o[8];
#pragma unroll
  for (int nt = 0; nt < 8; ++nt) o[nt] = z;
  float mrow[8], lrow[8];
#pragma unroll
  for (int v = 0; v < 8; ++v) { mrow[v] = -3.0e38f; lrow[v] = 0.f; }
  const float scale = 0.08838834764831845f;  // 1/sqrt(128)

  for (int j = 0; j <= qb; ++j) {
    __syncthreads();
    // K tile 64x128: async global->LDS (row-major direct copy)
#pragma unroll
    for (int c = t; c < 1024; c += 128) {
      int r = c >> 4, col = (c & 15) * 8;
      async_copy_b128(&lK[r][col],
                      &Kh[(size_t)(j * 64 + r) * (NKV * DH) + kvh * DH + col]);
    }
    // V tile transposed into lVt (VGPR path; overlaps the async K copies)
#pragma unroll
    for (int c = t; c < 1024; c += 128) {
      int r = c >> 4, col = (c & 15) * 8;
      v8h chunk =
          *(const v8h*)&Vh[(size_t)(j * 64 + r) * (NKV * DH) + kvh * DH + col];
#pragma unroll
      for (int e = 0; e < 8; ++e) lVt[col + e][r] = chunk[e];
    }
    wait_async0();
    __syncthreads();

    // Scores: S(16x64) = Q(16x128) * K^T, via 4 k-steps x 4 n-tiles
    v8f s[4];
#pragma unroll
    for (int nt = 0; nt < 4; ++nt) s[nt] = z;
#pragma unroll
    for (int kc = 0; kc < 4; ++kc) {
#pragma unroll
      for (int nt = 0; nt < 4; ++nt) {
        v16h bf = *(const v16h*)&lK[nt * 16 + lm][kc * 32 + lh * 16];
        s[nt] = wmma_f16(qf[kc], bf, s[nt]);
      }
    }
    // Scale + causal & padding mask
#pragma unroll
    for (int nt = 0; nt < 4; ++nt) {
      int kcol = j * 64 + nt * 16 + lm;
      bool kvalid = (amask[kcol] > 0);
#pragma unroll
      for (int v = 0; v < 8; ++v) {
        int qr = qrow0 + lh * 8 + v;
        float sv = s[nt][v] * scale;
        if (kcol > qr || !kvalid) sv = -3.0e38f;
        s[nt][v] = sv;
      }
    }
    // Online softmax (rows live across 16 lanes; reduce with shfl_xor)
    float pl[4][8];
#pragma unroll
    for (int v = 0; v < 8; ++v) {
      float tm = s[0][v];
#pragma unroll
      for (int nt = 1; nt < 4; ++nt) tm = fmaxf(tm, s[nt][v]);
#pragma unroll
      for (int off = 1; off < 16; off <<= 1)
        tm = fmaxf(tm, __shfl_xor(tm, off, 32));
      float mn = fmaxf(mrow[v], tm);
      float corr = __expf(mrow[v] - mn);
      mrow[v] = mn;
      lrow[v] *= corr;
#pragma unroll
      for (int nt = 0; nt < 8; ++nt) o[nt][v] *= corr;
      float rs = 0.f;
#pragma unroll
      for (int nt = 0; nt < 4; ++nt) {
        float p = __expf(s[nt][v] - mn);
        pl[nt][v] = p;
        rs += p;
      }
#pragma unroll
      for (int off = 1; off < 16; off <<= 1) rs += __shfl_xor(rs, off, 32);
      lrow[v] += rs;
    }
    // Stage P in LDS to re-layout into A-fragments
#pragma unroll
    for (int nt = 0; nt < 4; ++nt)
#pragma unroll
      for (int v = 0; v < 8; ++v)
        lP[wave][lh * 8 + v][nt * 16 + lm] = (_Float16)pl[nt][v];
    __syncthreads();

    // O += P(16x64) * V(64x128): 2 k-steps x 8 n-tiles
#pragma unroll
    for (int kc = 0; kc < 2; ++kc) {
      v16h pf = frag_a_row(&lP[wave][lm][0], kc * 32 + lh * 16);
#pragma unroll
      for (int nt = 0; nt < 8; ++nt) {
        v16h bf = *(const v16h*)&lVt[nt * 16 + lm][kc * 32 + lh * 16];
        o[nt] = wmma_f16(pf, bf, o[nt]);
      }
    }
  }

  // Normalize and write O as f16 [S, H, D]
#pragma unroll
  for (int v = 0; v < 8; ++v) {
    float inv = 1.0f / lrow[v];
    size_t row = (size_t)(qrow0 + lh * 8 + v) * (NH * DH) + h * DH;
#pragma unroll
    for (int nt = 0; nt < 8; ++nt)
      Oh[row + nt * 16 + lm] = (_Float16)(o[nt][v] * inv);
  }
}

// ---------------------------------------------------------------------------
// Host launcher
// ---------------------------------------------------------------------------
extern "C" void kernel_launch(void* const* d_in, const int* in_sizes, int n_in,
                              void* d_out, int out_size, void* d_ws,
                              size_t ws_size, hipStream_t stream) {
  const float* x  = (const float*)d_in[0];  // [1,2048,2048]
  const float* wq = (const float*)d_in[1];  // [2048,2048]
  const float* wk = (const float*)d_in[2];  // [2048,512]
  const float* wv = (const float*)d_in[3];  // [2048,512]
  const float* wo = (const float*)d_in[4];  // [2048,2048]
  const int* amask = (const int*)d_in[5];   // [1,2048]
  const int* pos   = (const int*)d_in[6];   // [1,2048]
  float* out = (float*)d_out;               // [1,2048,2048]

  const size_t M4 = 4u * 1024 * 1024;  // 2048*2048
  const size_t M1 = 1u * 1024 * 1024;  // 2048*512

  _Float16* Xh  = (_Float16*)d_ws;      // f16 activations
  _Float16* Wqh = Xh + M4;
  _Float16* Wkh = Wqh + M4;
  _Float16* Wvh = Wkh + M1;
  _Float16* Woh = Wvh + M1;
  float*    Qf  = (float*)(Woh + M4);   // fp32 pre-RoPE Q
  float*    Kf  = Qf + M4;              // fp32 pre-RoPE K
  _Float16* Qh  = (_Float16*)(Kf + M1);
  _Float16* Kh  = Qh + M4;
  _Float16* Vh  = Kh + M1;
  _Float16* Oh  = Vh + M1;              // f16 attention output

  // 1) fp32 -> f16 conversions
  cvt_f32_to_f16<<<(int)(M4 / 256), 256, 0, stream>>>(x, Xh, (int)M4);
  cvt_f32_to_f16<<<(int)(M4 / 256), 256, 0, stream>>>(wq, Wqh, (int)M4);
  cvt_f32_to_f16<<<(int)(M1 / 256), 256, 0, stream>>>(wk, Wkh, (int)M1);
  cvt_f32_to_f16<<<(int)(M1 / 256), 256, 0, stream>>>(wv, Wvh, (int)M1);
  cvt_f32_to_f16<<<(int)(M4 / 256), 256, 0, stream>>>(wo, Woh, (int)M4);

  // 2) QKV projections (WMMA); V goes straight to f16
  dim3 gq(HID / 128, S_LEN / 128);         // (16,16)
  dim3 gk((NKV * DH) / 128, S_LEN / 128);  // (4,16)
  gemm_f16<<<gq, 128, 0, stream>>>(Xh, Wqh, Qf, nullptr, S_LEN, NH * DH, HID);
  gemm_f16<<<gk, 128, 0, stream>>>(Xh, Wkh, Kf, nullptr, S_LEN, NKV * DH, HID);
  gemm_f16<<<gk, 128, 0, stream>>>(Xh, Wvh, nullptr, Vh, S_LEN, NKV * DH, HID);

  // 3) RoPE (fp32) -> f16 Q/K
  int qpairs = S_LEN * NH * (DH / 2);
  int kpairs = S_LEN * NKV * (DH / 2);
  rope_cvt<<<(qpairs + 255) / 256, 256, 0, stream>>>(Qf, pos, Qh, NH, qpairs);
  rope_cvt<<<(kpairs + 255) / 256, 256, 0, stream>>>(Kf, pos, Kh, NKV, kpairs);

  // 4) Flash attention (WMMA scores + WMMA PV, fp32 softmax)
  dim3 ga(S_LEN / 64, NH);  // (32,16)
  attn_flash<<<ga, 128, 0, stream>>>(Qh, Kh, Vh, amask, Oh);

  // 5) Output projection straight to fp32 d_out
  gemm_f16<<<gq, 128, 0, stream>>>(Oh, Woh, out, nullptr, S_LEN, HID, HID);
}
